// MambaDiffusion_51032801411305
// MI455X (gfx1250) — compile-verified, matrix operand order checked by reading, not run on hardware
//
#include <hip/hip_runtime.h>

// ---------------------------------------------------------------------------
// MambaDiffusion on gfx1250 (MI455X). L=1 collapses the scan; all heavy work
// is GEMMs driven through v_wmma_f32_16x16x32_bf16 with double-buffered LDS
// operand staging via GLOBAL_LOAD_ASYNC_TO_LDS_B128 (ASYNCcnt-tracked).
// Weights are converted to transposed bf16 each call (deterministic).
// ---------------------------------------------------------------------------

#define B_SZ 16384
#define IN_D 1024
#define HD   512
#define ID   1024
#define S_D  64
#define NL   8
#define CD   128
#define HD4  2048   // 4*HD
#define ID2  2048   // 2*ID

typedef __bf16 bf16;
typedef __attribute__((ext_vector_type(8)))  float v8f;
typedef __attribute__((ext_vector_type(8)))  bf16  v8bf;
typedef __attribute__((ext_vector_type(16))) bf16  v16bf;

// ---------------------------------------------------------------------------
// Weight convert + transpose: in f32 [K,N] row-major -> out bf16 [N,K].
// ---------------------------------------------------------------------------
__global__ __launch_bounds__(256) void kConvT(const float* __restrict__ in,
                                              bf16* __restrict__ out,
                                              int K, int N) {
  __shared__ float tile[32][33];
  int n0 = blockIdx.x * 32;
  int k0 = blockIdx.y * 32;
  int tx = threadIdx.x;        // 0..31
  int ty = threadIdx.y;        // 0..7
  #pragma unroll
  for (int r = ty; r < 32; r += 8)
    tile[r][tx] = in[(size_t)(k0 + r) * N + (n0 + tx)];
  __syncthreads();
  #pragma unroll
  for (int r = ty; r < 32; r += 8)
    out[(size_t)(n0 + r) * K + (k0 + tx)] = (bf16)tile[tx][r];
}

// f32 -> bf16 elementwise cast
__global__ __launch_bounds__(256) void kCast(const float* __restrict__ in,
                                             bf16* __restrict__ out, size_t n) {
  size_t i = (size_t)blockIdx.x * 256 + threadIdx.x;
  if (i < n) out[i] = (bf16)in[i];
}

// dt bias combine: out[j] = dt_b[j] + dt_bias[j]
__global__ void kBias2(const float* __restrict__ a, const float* __restrict__ b,
                       float* __restrict__ out, int n) {
  int i = blockIdx.x * 256 + threadIdx.x;
  if (i < n) out[i] = a[i] + b[i];
}

// ---------------------------------------------------------------------------
// Time embedding: [cos(t*f_j), sin(t*f_j)], f_j = exp(-ln(1e4)*j/256), bf16.
// ---------------------------------------------------------------------------
__global__ __launch_bounds__(256) void kTimeEmb(const int* __restrict__ ts,
                                                bf16* __restrict__ out) {
  size_t i = (size_t)blockIdx.x * 256 + threadIdx.x;
  if (i >= (size_t)B_SZ * 256) return;
  int j = (int)(i & 255);
  size_t row = i >> 8;
  float f = __expf(-9.210340371976184f * (float)j * (1.0f / 256.0f));
  float a = (float)ts[row] * f;
  out[row * HD + j]        = (bf16)__cosf(a);
  out[row * HD + 256 + j]  = (bf16)__sinf(a);
}

// ---------------------------------------------------------------------------
// Row LayerNorm over `cols` (eps 1e-5), result stored bf16 for next GEMM.
// ---------------------------------------------------------------------------
__global__ __launch_bounds__(256) void kLayerNorm(const float* __restrict__ x,
                                                  const float* __restrict__ w,
                                                  const float* __restrict__ b,
                                                  bf16* __restrict__ out,
                                                  int cols) {
  __shared__ float s[256];
  int row = blockIdx.x;
  const float* p = x + (size_t)row * cols;
  int tid = threadIdx.x;
  float acc = 0.f;
  for (int j = tid; j < cols; j += 256) acc += p[j];
  s[tid] = acc; __syncthreads();
  for (int st = 128; st > 0; st >>= 1) {
    if (tid < st) s[tid] += s[tid + st];
    __syncthreads();
  }
  float mean = s[0] / (float)cols;
  __syncthreads();
  acc = 0.f;
  for (int j = tid; j < cols; j += 256) { float d = p[j] - mean; acc += d * d; }
  s[tid] = acc; __syncthreads();
  for (int st = 128; st > 0; st >>= 1) {
    if (tid < st) s[tid] += s[tid + st];
    __syncthreads();
  }
  float inv = rsqrtf(s[0] / (float)cols + 1e-5f);
  for (int j = tid; j < cols; j += 256)
    out[(size_t)row * cols + j] = (bf16)((p[j] - mean) * inv * w[j] + b[j]);
}

// ---------------------------------------------------------------------------
// WMMA bf16 GEMM with async-LDS double buffering.
//   A  [M,K] bf16 row-major, Wt [N,K] bf16 (pre-transposed).
//   Block = 128 threads (4 waves) -> 128(M) x 64(N) tile.
//   Per 32-K step: stage A(128x32) + B(64x32) tiles into LDS with
//   GLOBAL_LOAD_ASYNC_TO_LDS_B128 (6 issues/wave), overlap with WMMA on the
//   other buffer, sync with s_wait_asynccnt + workgroup barrier.
//   Each wave computes 32x64 (8 accumulators, 8 wmma / k-step).
// Epilogue: +bias, act (0 none, 1 silu, 2 softplus), +resid, f32/bf16 stores.
// Requires M%128==0, N%64==0, K%32==0.
// ---------------------------------------------------------------------------
#define LDS_STRIDE 40   // bf16 elems per row (80 B): conflict-free ds_load_b128

__device__ __forceinline__ void asyncCopy16(unsigned ldsOff,
                                            const bf16* __restrict__ g) {
  asm volatile("global_load_async_to_lds_b128 %0, %1, off"
               :: "v"(ldsOff), "v"(g)
               : "memory");
}

__device__ __forceinline__ unsigned ldsAddr(const bf16* p) {
  return (unsigned)(unsigned long long)p;   // low 32 bits = LDS byte offset
}

__device__ __forceinline__ v16bf fragLds(const bf16* __restrict__ row,
                                         int laneHi) {
  // elements 0..7  <- K = laneHi*8 + [0..7]
  // elements 8..15 <- K = 16 + laneHi*8 + [0..7]
  v8bf lo = *(const v8bf*)(row + laneHi * 8);
  v8bf hi = *(const v8bf*)(row + 16 + laneHi * 8);
  return __builtin_shufflevector(lo, hi, 0, 1, 2, 3, 4, 5, 6, 7,
                                         8, 9, 10, 11, 12, 13, 14, 15);
}

__global__ __launch_bounds__(128) void kGemm(const bf16* __restrict__ A,
                                             const bf16* __restrict__ Wt,
                                             const float* __restrict__ bias,
                                             const float* __restrict__ resid,
                                             float* __restrict__ outF,
                                             bf16* __restrict__ outB,
                                             int M, int N, int K, int act) {
  __shared__ bf16 ldsA[2][128 * LDS_STRIDE];   // 2 x 10240 B
  __shared__ bf16 ldsB[2][64 * LDS_STRIDE];    // 2 x  5120 B

  int tid    = threadIdx.x;
  int lane   = tid & 31;
  int wave   = tid >> 5;
  int laneLo = lane & 15;
  int laneHi = lane >> 4;
  int m0 = blockIdx.y * 128;
  int n0 = blockIdx.x * 64;

  // Issue one 32-K-deep tile pair into LDS buffer `buf`.
  auto issue = [&](int buf, int k0) {
    #pragma unroll
    for (int i = 0; i < 4; ++i) {            // A: 512 chunks of 16 B
      int c = tid + i * 128;
      int row = c >> 2, k8 = (c & 3) * 8;
      asyncCopy16(ldsAddr(&ldsA[buf][row * LDS_STRIDE + k8]),
                  A + (size_t)(m0 + row) * K + k0 + k8);
    }
    #pragma unroll
    for (int i = 0; i < 2; ++i) {            // B: 256 chunks of 16 B
      int c = tid + i * 128;
      int col = c >> 2, k8 = (c & 3) * 8;
      asyncCopy16(ldsAddr(&ldsB[buf][col * LDS_STRIDE + k8]),
                  Wt + (size_t)(n0 + col) * K + k0 + k8);
    }
  };

  v8f acc[8];
  #pragma unroll
  for (int t = 0; t < 8; ++t)
    acc[t] = (v8f){0.f, 0.f, 0.f, 0.f, 0.f, 0.f, 0.f, 0.f};

  issue(0, 0);
  for (int k0 = 0; k0 < K; k0 += 32) {
    int cur = (k0 >> 5) & 1;
    if (k0 + 32 < K) {
      issue(1 - cur, k0 + 32);
      asm volatile("s_wait_asynccnt 0x6" ::: "memory");  // prev 6 complete
    } else {
      asm volatile("s_wait_asynccnt 0x0" ::: "memory");
    }
    __syncthreads();                                     // tiles visible to all

    const bf16* aP = &ldsA[cur][(wave * 32 + laneLo) * LDS_STRIDE];
    const bf16* bP = &ldsB[cur][laneLo * LDS_STRIDE];
    v16bf a0 = fragLds(aP, laneHi);
    v16bf a1 = fragLds(aP + 16 * LDS_STRIDE, laneHi);
    #pragma unroll
    for (int t = 0; t < 4; ++t) {
      v16bf w = fragLds(bP + t * 16 * LDS_STRIDE, laneHi);
      acc[t]     = __builtin_amdgcn_wmma_f32_16x16x32_bf16(
                       false, a0, false, w, (short)0, acc[t],     false, false);
      acc[4 + t] = __builtin_amdgcn_wmma_f32_16x16x32_bf16(
                       false, a1, false, w, (short)0, acc[4 + t], false, false);
    }
    __syncthreads();   // all waves done reading before buffer is rewritten
  }

  #pragma unroll
  for (int s = 0; s < 8; ++s) {
    int t  = s & 3;
    int ms = wave * 32 + ((s >> 2) ? 16 : 0);
    int rowBase = m0 + ms + laneHi * 8;
    int col = n0 + t * 16 + laneLo;
    float bv = bias ? bias[col] : 0.f;
    #pragma unroll
    for (int r = 0; r < 8; ++r) {
      float v = acc[s][r] + bv;
      if (act == 1) {                       // silu
        v = v / (1.f + __expf(-v));
      } else if (act == 2) {                // softplus
        v = (v > 20.f) ? v : log1pf(__expf(v));
      }
      size_t idx = (size_t)(rowBase + r) * N + col;
      if (resid) v += resid[idx];
      if (outF) outF[idx] = v;
      if (outB) outB[idx] = (bf16)v;
    }
  }
}

// ---------------------------------------------------------------------------
// Fuse: conv-center + bias, SiLU, add cond projection; store f32 + bf16.
// ---------------------------------------------------------------------------
__global__ __launch_bounds__(256) void kFuseConvCond(
    const float* __restrict__ xp,     // [B, 2*ID] (first ID = x_proj)
    const float* __restrict__ cp,     // [B, ID]
    const float* __restrict__ convw,  // [ID,1,3] for this layer
    const float* __restrict__ convb,  // [ID]
    float* __restrict__ xpf, bf16* __restrict__ xpb) {
  size_t i = (size_t)blockIdx.x * 256 + threadIdx.x;
  if (i >= (size_t)B_SZ * ID) return;
  int    j   = (int)(i & (ID - 1));
  size_t row = i >> 10;
  float v = xp[row * ID2 + j] * convw[j * 3 + 1] + convb[j];
  v = v / (1.f + __expf(-v));           // silu
  v += cp[i];
  xpf[i] = v;
  xpb[i] = (bf16)v;
}

// Row mean over `cols` with scale (scale = 1/cols)
__global__ __launch_bounds__(256) void kRowMean(const float* __restrict__ in,
                                                float* __restrict__ out,
                                                int cols, float scale) {
  __shared__ float s[256];
  int row = blockIdx.x;
  const float* p = in + (size_t)row * cols;
  int tid = threadIdx.x;
  float acc = 0.f;
  for (int j = tid; j < cols; j += 256) acc += p[j];
  s[tid] = acc; __syncthreads();
  for (int st = 128; st > 0; st >>= 1) {
    if (tid < st) s[tid] += s[tid + st];
    __syncthreads();
  }
  if (tid == 0) out[row] = s[0] * scale;
}

// y_scalar[b] = dtm[b] * xmean[b] * dot(Bp[b,:], Cp[b,:])   (S_D = 64)
__global__ __launch_bounds__(256) void kYs(const float* __restrict__ Bp,
                                           const float* __restrict__ Cp,
                                           const float* __restrict__ dtm,
                                           const float* __restrict__ xmean,
                                           float* __restrict__ ys) {
  __shared__ float s[256];
  int tid = threadIdx.x;
  int sub = tid & 63;                         // lane within row
  int row = blockIdx.x * 4 + (tid >> 6);      // 4 rows per block
  size_t off = (size_t)row * S_D + sub;
  s[tid] = Bp[off] * Cp[off];
  __syncthreads();
  for (int st = 32; st > 0; st >>= 1) {
    if (sub < st) s[tid] += s[tid + st];
    __syncthreads();
  }
  if (sub == 0) ys[row] = dtm[row] * xmean[row] * s[tid];
}

// y = (ys[b] + x_proj*Dp) * silu(gate); gate = xp[:, ID:2*ID]. Output bf16.
__global__ __launch_bounds__(256) void kGate(const float* __restrict__ xp,
                                             const float* __restrict__ xpf,
                                             const float* __restrict__ Dp,
                                             const float* __restrict__ ys,
                                             bf16* __restrict__ yb) {
  size_t i = (size_t)blockIdx.x * 256 + threadIdx.x;
  if (i >= (size_t)B_SZ * ID) return;
  int    j   = (int)(i & (ID - 1));
  size_t row = i >> 10;
  float g = xp[row * ID2 + ID + j];
  float sg = g / (1.f + __expf(-g));
  yb[i] = (bf16)((ys[row] + xpf[i] * Dp[j]) * sg);
}

// ---------------------------------------------------------------------------
// Host-side orchestration
// ---------------------------------------------------------------------------
static inline size_t alignUp(size_t x) { return (x + 255) & ~(size_t)255; }

extern "C" void kernel_launch(void* const* d_in, const int* in_sizes, int n_in,
                              void* d_out, int out_size, void* d_ws, size_t ws_size,
                              hipStream_t stream) {
  (void)in_sizes; (void)n_in; (void)out_size; (void)ws_size;

  const float* x        = (const float*)d_in[0];
  const int*   timestep = (const int*)  d_in[1];
  const float* cond_in  = (const float*)d_in[2];
  const float* in_w     = (const float*)d_in[3];
  const float* in_b     = (const float*)d_in[4];
  const float* t_w1     = (const float*)d_in[5];
  const float* t_b1     = (const float*)d_in[6];
  const float* t_w2     = (const float*)d_in[7];
  const float* t_b2     = (const float*)d_in[8];
  const float* c_w1     = (const float*)d_in[9];
  const float* c_b1     = (const float*)d_in[10];
  const float* c_w2     = (const float*)d_in[11];
  const float* c_b2     = (const float*)d_in[12];
  const float* norm_w   = (const float*)d_in[13];
  const float* norm_b   = (const float*)d_in[14];
  const float* inproj_w = (const float*)d_in[15];
  const float* inproj_b = (const float*)d_in[16];
  const float* conv_w   = (const float*)d_in[17];
  const float* conv_b   = (const float*)d_in[18];
  const float* dt_w     = (const float*)d_in[19];
  const float* dt_b     = (const float*)d_in[20];
  const float* dt_bias  = (const float*)d_in[21];
  /* d_in[22] = A_log : dead with L=1, h0=0 */
  const float* Dp       = (const float*)d_in[23];
  const float* B_w      = (const float*)d_in[24];
  const float* B_b      = (const float*)d_in[25];
  const float* C_w      = (const float*)d_in[26];
  const float* C_b      = (const float*)d_in[27];
  const float* condp_w  = (const float*)d_in[28];
  const float* condp_b  = (const float*)d_in[29];
  const float* out_w    = (const float*)d_in[30];
  const float* out_b    = (const float*)d_in[31];
  const float* on_w     = (const float*)d_in[32];
  const float* on_b     = (const float*)d_in[33];
  const float* op_w     = (const float*)d_in[34];
  const float* op_b     = (const float*)d_in[35];

  // ---- workspace layout ----
  char* ws = (char*)d_ws;
  size_t cur = 0;
  auto takeB = [&](size_t elems) { bf16*  p = (bf16*)(ws + cur);  cur = alignUp(cur + elems * sizeof(bf16));  return p; };
  auto takeF = [&](size_t elems) { float* p = (float*)(ws + cur); cur = alignUp(cur + elems * sizeof(float)); return p; };

  // transposed bf16 weights [N,K]
  bf16* in_wt    = takeB((size_t)HD * IN_D);
  bf16* t_w1t    = takeB((size_t)HD4 * HD);
  bf16* t_w2t    = takeB((size_t)HD * HD4);
  bf16* c_w1t    = takeB((size_t)HD * CD);
  bf16* c_w2t    = takeB((size_t)HD * HD);
  bf16* inproj_t = takeB((size_t)NL * ID2 * HD);
  bf16* dt_wt    = takeB((size_t)NL * ID * ID);
  bf16* B_wt     = takeB((size_t)NL * S_D * ID);
  bf16* C_wt     = takeB((size_t)NL * S_D * ID);
  bf16* condp_t  = takeB((size_t)NL * ID * HD);
  bf16* out_wt   = takeB((size_t)NL * HD * ID);
  bf16* op_wt    = takeB((size_t)IN_D * HD);
  float* dtb2    = takeF((size_t)NL * ID);

  // activations
  bf16*  xb    = takeB((size_t)B_SZ * IN_D);
  bf16*  cb128 = takeB((size_t)B_SZ * CD);
  bf16*  t0b   = takeB((size_t)B_SZ * HD);
  bf16*  thb   = takeB((size_t)B_SZ * HD4);
  bf16*  chb   = takeB((size_t)B_SZ * HD);
  bf16*  condb = takeB((size_t)B_SZ * HD);
  bf16*  hnb   = takeB((size_t)B_SZ * HD);
  bf16*  xpb   = takeB((size_t)B_SZ * ID);
  bf16*  yb    = takeB((size_t)B_SZ * ID);
  float* h     = takeF((size_t)B_SZ * HD);
  float* condf = takeF((size_t)B_SZ * HD);
  float* xp    = takeF((size_t)B_SZ * ID2);
  float* cp    = takeF((size_t)B_SZ * ID);
  float* xpf   = takeF((size_t)B_SZ * ID);
  float* dtf   = takeF((size_t)B_SZ * ID);
  float* BpA   = takeF((size_t)B_SZ * S_D);
  float* CpA   = takeF((size_t)B_SZ * S_D);
  float* dtm   = takeF((size_t)B_SZ);
  float* xmean = takeF((size_t)B_SZ);
  float* ysv   = takeF((size_t)B_SZ);

  dim3 ctb(32, 8);
  auto CT = [&](const float* src, bf16* dst, int K, int N) {
    kConvT<<<dim3(N / 32, K / 32), ctb, 0, stream>>>(src, dst, K, N);
  };
  auto GEMM = [&](const bf16* A, const bf16* Wt, const float* bias,
                  const float* resid, float* oF, bf16* oB,
                  int M, int N, int K, int act) {
    kGemm<<<dim3(N / 64, M / 128), 128, 0, stream>>>(A, Wt, bias, resid, oF, oB,
                                                     M, N, K, act);
  };

  // ---- weight conversion (runs every call; deterministic) ----
  CT(in_w,  in_wt, IN_D, HD);
  CT(t_w1,  t_w1t, HD,  HD4);
  CT(t_w2,  t_w2t, HD4, HD);
  CT(c_w1,  c_w1t, CD,  HD);
  CT(c_w2,  c_w2t, HD,  HD);
  CT(op_w,  op_wt, HD,  IN_D);
  for (int l = 0; l < NL; ++l) {
    CT(inproj_w + (size_t)l * HD * ID2, inproj_t + (size_t)l * ID2 * HD, HD, ID2);
    CT(dt_w    + (size_t)l * ID * ID,   dt_wt    + (size_t)l * ID * ID,  ID, ID);
    CT(B_w     + (size_t)l * ID * S_D,  B_wt     + (size_t)l * S_D * ID, ID, S_D);
    CT(C_w     + (size_t)l * ID * S_D,  C_wt     + (size_t)l * S_D * ID, ID, S_D);
    CT(condp_w + (size_t)l * HD * ID,   condp_t  + (size_t)l * ID * HD,  HD, ID);
    CT(out_w   + (size_t)l * ID * HD,   out_wt   + (size_t)l * HD * ID,  ID, HD);
    kBias2<<<(ID + 255) / 256, 256, 0, stream>>>(dt_b + (size_t)l * ID,
                                                 dt_bias + (size_t)l * ID,
                                                 dtb2 + (size_t)l * ID, ID);
  }

  // ---- input casts ----
  {
    size_t n = (size_t)B_SZ * IN_D;
    kCast<<<(unsigned)((n + 255) / 256), 256, 0, stream>>>(x, xb, n);
    size_t m = (size_t)B_SZ * CD;
    kCast<<<(unsigned)((m + 255) / 256), 256, 0, stream>>>(cond_in, cb128, m);
  }

  // ---- stem: h = x @ in_w + in_b ----
  GEMM(xb, in_wt, in_b, nullptr, h, nullptr, B_SZ, HD, IN_D, 0);

  // ---- conditioning: temb + cemb ----
  kTimeEmb<<<(B_SZ * 256) / 256, 256, 0, stream>>>(timestep, t0b);
  GEMM(t0b, t_w1t, t_b1, nullptr, nullptr, thb, B_SZ, HD4, HD, 1);    // silu
  GEMM(thb, t_w2t, t_b2, nullptr, condf, nullptr, B_SZ, HD, HD4, 0);  // temb
  GEMM(cb128, c_w1t, c_b1, nullptr, nullptr, chb, B_SZ, HD, CD, 1);   // silu
  GEMM(chb, c_w2t, c_b2, condf, condf, condb, B_SZ, HD, HD, 0);       // + temb

  // ---- layers ----
  const unsigned ewGrid = (unsigned)(((size_t)B_SZ * ID + 255) / 256);
  for (int l = 0; l < NL; ++l) {
    kLayerNorm<<<B_SZ, 256, 0, stream>>>(h, norm_w + (size_t)l * HD,
                                         norm_b + (size_t)l * HD, hnb, HD);
    GEMM(hnb, inproj_t + (size_t)l * ID2 * HD, inproj_b + (size_t)l * ID2,
         nullptr, xp, nullptr, B_SZ, ID2, HD, 0);
    GEMM(condb, condp_t + (size_t)l * ID * HD, condp_b + (size_t)l * ID,
         nullptr, cp, nullptr, B_SZ, ID, HD, 0);
    kFuseConvCond<<<ewGrid, 256, 0, stream>>>(xp, cp,
        conv_w + (size_t)l * ID * 3, conv_b + (size_t)l * ID, xpf, xpb);
    kRowMean<<<B_SZ, 256, 0, stream>>>(xpf, xmean, ID, 1.0f / ID);
    GEMM(xpb, dt_wt + (size_t)l * ID * ID, dtb2 + (size_t)l * ID,
         nullptr, dtf, nullptr, B_SZ, ID, ID, 2);                      // softplus
    kRowMean<<<B_SZ, 256, 0, stream>>>(dtf, dtm, ID, 1.0f / ID);
    GEMM(xpb, B_wt + (size_t)l * S_D * ID, B_b + (size_t)l * S_D,
         nullptr, BpA, nullptr, B_SZ, S_D, ID, 0);
    GEMM(xpb, C_wt + (size_t)l * S_D * ID, C_b + (size_t)l * S_D,
         nullptr, CpA, nullptr, B_SZ, S_D, ID, 0);
    kYs<<<B_SZ / 4, 256, 0, stream>>>(BpA, CpA, dtm, xmean, ysv);
    kGate<<<ewGrid, 256, 0, stream>>>(xp, xpf, Dp + (size_t)l * ID, ysv, yb);
    GEMM(yb, out_wt + (size_t)l * HD * ID, out_b + (size_t)l * HD,
         h, h, nullptr, B_SZ, HD, ID, 0);                              // residual
  }

  // ---- head ----
  kLayerNorm<<<B_SZ, 256, 0, stream>>>(h, on_w, on_b, hnb, HD);
  GEMM(hnb, op_wt, op_b, nullptr, (float*)d_out, nullptr, B_SZ, IN_D, HD, 0);
}
